// Hydra_16372415332784
// MI455X (gfx1250) — matrix operand order repaired and checked by reading, not compiled, not run on hardware
//
#include <hip/hip_runtime.h>
#include <hip/hip_bf16.h>
#include <stdint.h>

// ---------------------------------------------------------------------------
// Hydra (bidirectional Mamba-2) forward for MI455X / gfx1250.
// HBM-bound (~0.9GB traffic @ 23.3TB/s ~= 40us; ~80 GFLOP << WMMA ceiling).
// All contractions -> v_wmma_f32_16x16x32_bf16; weights + SSD activations
// stored bf16 in workspace; bf16 2D tiles (GEMM weights, SSD x/B/prev) are
// DMA'd by the Tensor Data Mover (tensor_load_to_lds), double-buffered and
// paced with s_wait_tensorcnt so DMA overlaps WMMA.
// ---------------------------------------------------------------------------

typedef __bf16 bf16_t;
typedef __attribute__((ext_vector_type(16))) __bf16 v16bf;
typedef __attribute__((ext_vector_type(4)))  __bf16 bf16x4;
typedef __attribute__((ext_vector_type(8)))  float  v8f;
typedef __attribute__((ext_vector_type(4)))  unsigned int u32x4;
typedef __attribute__((ext_vector_type(8)))  int i32x8;
typedef __attribute__((ext_vector_type(4)))  int i32x4;

#define D_MODEL   1024
#define D_INNER   2048
#define HEADDIM   64
#define NHEADS    32
#define D_STATE   64
#define D_CONV    7
#define CHUNK     256
#define CONV_DIM  2304
#define D_IN_PROJ 4416
#define B_SZ      2
#define SEQLEN    2048
#define NCHUNKS   (SEQLEN / CHUNK)   // 8
#define BB        (2 * B_SZ)         // 4
#define MTOT      (B_SZ * SEQLEN)    // 4096

#if defined(__has_builtin)
#if __has_builtin(__builtin_amdgcn_tensor_load_to_lds)
#define HAVE_TDM 1
#endif
#endif
#ifndef HAVE_TDM
#define HAVE_TDM 0
#endif

// -------------------------------- helpers ----------------------------------

__device__ __forceinline__ v8f v8f_zero() {
  v8f z;
#pragma unroll
  for (int i = 0; i < 8; ++i) z[i] = 0.0f;
  return z;
}

__device__ __forceinline__ v8f wmma_bf16(v16bf a, v16bf b, v8f c) {
  return __builtin_amdgcn_wmma_f32_16x16x32_bf16(false, a, false, b,
                                                 (short)0, c, false, false);
}

// A-matrix 16x32 fragment: lane (g,r) holds row M=r;
// halves[j] = K = g*16+j  (one 32B run).
__device__ __forceinline__ v16bf load_a_frag(const bf16_t* base, int ld,
                                             int r, int g) {
  const bf16_t* p = base + (size_t)r * ld + g * 16;
  v16bf a;
#pragma unroll
  for (int j = 0; j < 16; ++j) a[j] = p[j];
  return a;
}

// B-matrix 32x16 fragment, storage row-per-column (base[n*ld + k]):
// lane column N=r, halves[j] = B[g*16+j][r]  (one 32B run).
__device__ __forceinline__ v16bf load_b_frag_rowN(const bf16_t* base, int ld,
                                                  int r, int g) {
  const bf16_t* p = base + (size_t)r * ld + g * 16;
  v16bf b;
#pragma unroll
  for (int j = 0; j < 16; ++j) b[j] = p[j];
  return b;
}

// B-matrix fragment, storage K-major (base[k*ld + n]).
__device__ __forceinline__ v16bf load_b_frag_rowK(const bf16_t* base, int ld,
                                                  int n, int g) {
  const bf16_t* p = base + (size_t)(g * 16) * ld + n;
  v16bf b;
#pragma unroll
  for (int j = 0; j < 16; ++j) b[j] = p[j * ld];
  return b;
}

__device__ __forceinline__ unsigned rfl(unsigned x) {
  return (unsigned)__builtin_amdgcn_readfirstlane((int)x);
}

#if HAVE_TDM
// TDM 2D bf16 tile load into LDS with row padding. D# per ISA ch.8:
// group0 {count=1, lds_addr, global_addr, type=2}; group1 {data_size=1(2B),
// pad_enable, pad_interval (enum: 3=16DW rows, 4=32DW rows), pad_amount
// (enc 3 = 4 DWORDs), tensor_dim0/1 (OOB clamp), tile_w x tile_h,
// dim0_stride}. Row padding reproduces the LDS strides the fragment
// loaders expect (40 halves for 32-wide tiles, 72 halves for 64-wide).
__device__ __forceinline__ void tdm_load_2d(unsigned lds_addr,
                                            const bf16_t* gptr,
                                            unsigned k_rem, unsigned n_rem,
                                            unsigned ld, unsigned tile_w,
                                            unsigned tile_h, unsigned pad_int,
                                            unsigned pad_amt) {
  unsigned long long ga = (unsigned long long)(size_t)(const void*)gptr;
  u32x4 g0;
  g0[0] = rfl(1u);                                   // count=1, user mode
  g0[1] = rfl(lds_addr);
  g0[2] = rfl((unsigned)ga);
  g0[3] = rfl((unsigned)((ga >> 32) & 0x01FFFFFFu) | (2u << 30)); // type=2
  i32x8 g1;
  g1[0] = (int)rfl((1u << 16) | (1u << 20) | (pad_int << 22) |
                   (pad_amt << 25));
  g1[1] = (int)rfl((k_rem & 0xFFFFu) << 16);         // tensor_dim0 lo16
  g1[2] = (int)rfl(((k_rem >> 16) & 0xFFFFu) | ((n_rem & 0xFFFFu) << 16));
  g1[3] = (int)rfl(((n_rem >> 16) & 0xFFFFu) | (tile_w << 16));
  g1[4] = (int)rfl(tile_h);                          // tile_dim1 (dim2=0)
  g1[5] = (int)rfl(ld);                              // dim0_stride lo32
  g1[6] = 0;
  g1[7] = 0;
  i32x4 g2; g2[0] = 0; g2[1] = 0; g2[2] = 0; g2[3] = 0;
  i32x4 g3 = g2;
#if defined(__clang_major__) && __clang_major__ >= 23
  i32x8 g4;
#pragma unroll
  for (int i = 0; i < 8; ++i) g4[i] = 0;
  __builtin_amdgcn_tensor_load_to_lds(g0, g1, g2, g3, g4, 0);
#else
  __builtin_amdgcn_tensor_load_to_lds(g0, g1, g2, g3, 0);
#endif
}
#endif  // HAVE_TDM

// ------------------------- K1: generic WMMA GEMM ---------------------------
// C[M,N] = A[M,K] (f32) * W[N,K]^T (bf16), f32 out. Block 256 thr = 8 waves;
// tile 128(M) x 64(N); K step 32. M % 128 == 0, K % 32 == 0 (all call sites).
// W tiles double-buffered via TDM; A tiles staged f32->bf16 with b128 loads
// (gather-then-convert so 4 loads stay in flight per thread).

#define GT_M 128
#define GT_N 64
#define GT_K 32
#define LDA_S 40   // halves; 80B rows -> 16B-aligned fragment runs
#define LDB_S 40

__global__ __launch_bounds__(256) void gemm_wmma_kernel(
    const float* __restrict__ A, const bf16_t* __restrict__ W,
    float* __restrict__ C, int M, int N, int K, int lda, int ldw, int ldc) {
  __shared__ __align__(16) bf16_t As[GT_M * LDA_S];
  __shared__ __align__(16) bf16_t Ws[2][GT_N * LDB_S];
  const int tid  = threadIdx.x;
  const int wave = tid >> 5;
  const int lane = tid & 31;
  const int g = lane >> 4, r = lane & 15;
  const int m0 = blockIdx.y * GT_M;
  const int n0 = blockIdx.x * GT_N;

  v8f acc[4];
#pragma unroll
  for (int i = 0; i < 4; ++i) acc[i] = v8f_zero();

#if HAVE_TDM
  if (wave == 0)
    tdm_load_2d((unsigned)(size_t)(void*)&Ws[0][0], W + (size_t)n0 * ldw,
                (unsigned)K, (unsigned)(N - n0), (unsigned)ldw,
                32u, 64u, 3u, 3u);
#endif

  for (int k0 = 0; k0 < K; k0 += GT_K) {
    const int buf = (k0 >> 5) & 1;
    // Stage A tile 128x32 f32 -> bf16: gather first (4 b128 loads in
    // flight), then convert + ds_store.
    float4 va[4];
#pragma unroll
    for (int it = 0; it < 4; ++it) {
      int cid = tid + it * 256;          // 1024 float4 chunks
      int row = cid >> 3, c4 = cid & 7;
      va[it] = *(const float4*)(A + (size_t)(m0 + row) * lda + k0 + c4 * 4);
    }
#pragma unroll
    for (int it = 0; it < 4; ++it) {
      int cid = tid + it * 256;
      int row = cid >> 3, c4 = cid & 7;
      bf16_t* d = &As[row * LDA_S + c4 * 4];
      d[0] = (bf16_t)va[it].x; d[1] = (bf16_t)va[it].y;
      d[2] = (bf16_t)va[it].z; d[3] = (bf16_t)va[it].w;
    }
#if HAVE_TDM
    if (wave == 0) {
      __builtin_amdgcn_s_wait_tensorcnt((short)0);   // Ws[buf] ready
      if (k0 + GT_K < K)                             // next tile DMA
        tdm_load_2d((unsigned)(size_t)(void*)&Ws[buf ^ 1][0],
                    W + (size_t)n0 * ldw + k0 + GT_K,
                    (unsigned)(K - k0 - GT_K), (unsigned)(N - n0),
                    (unsigned)ldw, 32u, 64u, 3u, 3u);
    }
#else
    // Fallback: cooperative bf16 tile copy with row guard.
#pragma unroll
    for (int it = 0; it < 2; ++it) {
      int cid = tid + it * 256;          // 512 bf16x4 chunks
      int row = cid >> 3, c4 = cid & 7;
      int gn = n0 + row;
      bf16x4 v;
      if (gn < N) {
        v = *(const bf16x4*)(W + (size_t)gn * ldw + k0 + c4 * 4);
      } else {
        v[0] = v[1] = v[2] = v[3] = (bf16_t)0.0f;
      }
      *(bf16x4*)&Ws[buf][row * LDB_S + c4 * 4] = v;
    }
#endif
    __syncthreads();

    if (k0 + GT_K < K) {                 // prefetch next A slab (CDNA5)
      int row = tid >> 1;
      __builtin_prefetch(A + (size_t)(m0 + row) * lda + k0 + GT_K +
                             (tid & 1) * 16, 0, 1);
    }

    v16bf af = load_a_frag(&As[(wave * 16) * LDA_S], LDA_S, r, g);
#pragma unroll
    for (int nt = 0; nt < 4; ++nt) {
      v16bf bfr = load_b_frag_rowN(&Ws[buf][(nt * 16) * LDB_S], LDB_S, r, g);
      acc[nt] = wmma_bf16(af, bfr, acc[nt]);
    }
    __syncthreads();
  }

  // D layout: lane (g,r), VGPR v -> (m = v + 8g, n = r).
  const int mrow = m0 + wave * 16 + 8 * g;
#pragma unroll
  for (int nt = 0; nt < 4; ++nt) {
    int col = n0 + nt * 16 + r;
    if (col < N) {
#pragma unroll
      for (int v = 0; v < 8; ++v)
        C[(size_t)(mrow + v) * ldc + col] = acc[nt][v];
    }
  }
}

// ----------------------- K0: f32 -> bf16 converter -------------------------
__global__ __launch_bounds__(256) void cvt_bf16_kernel(
    const float* __restrict__ src, bf16_t* __restrict__ dst, int n) {
  for (int i = blockIdx.x * 256 + threadIdx.x; i < n; i += gridDim.x * 256)
    dst[i] = (bf16_t)src[i];
}

// --------------------- K2: dt softplus + per-chunk cumsum ------------------
__global__ void dtprep_kernel(const float* __restrict__ zxbcdt,
                              const float* __restrict__ dt_bias,
                              const float* __restrict__ A_log,
                              float* __restrict__ dt_out,
                              float* __restrict__ dAcs_out,
                              float* __restrict__ dAlast) {
  const int blk = blockIdx.x;
  const int bb = blk / NCHUNKS;
  const int c  = blk % NCHUNKS;
  const int h  = threadIdx.x;
  const int b  = bb % B_SZ;
  const bool bw = (bb >= B_SZ);
  const float Aneg = -__expf(A_log[h]);
  const float bias = dt_bias[h];
  float cs = 0.0f;
  for (int t = 0; t < CHUNK; ++t) {
    int l    = c * CHUNK + t;
    int lsrc = bw ? (SEQLEN - 1 - l) : l;
    int hsrc = bw ? (NHEADS + h) : h;
    float raw = zxbcdt[((size_t)b * SEQLEN + lsrc) * D_IN_PROJ +
                       (D_INNER + CONV_DIM) + hsrc];
    float x  = raw + bias;
    float dt = (x > 20.0f) ? x : log1pf(__expf(x));
    cs += dt * Aneg;
    dt_out[((size_t)bb * SEQLEN + l) * NHEADS + h]   = dt;
    dAcs_out[((size_t)bb * SEQLEN + l) * NHEADS + h] = cs;
  }
  dAlast[(bb * NCHUNKS + c) * NHEADS + h] = cs;
}

// ---------------------- K3: depthwise conv7 + SiLU -------------------------
__global__ __launch_bounds__(256) void dwconv_kernel(
    const float* __restrict__ zxbcdt, const float* __restrict__ conv_w,
    const float* __restrict__ conv_b, float* __restrict__ convout) {
  const int ch = blockIdx.x * 256 + threadIdx.x;
  const int bl = blockIdx.y;
  const int b = bl / SEQLEN, l = bl % SEQLEN;
  const float* src = zxbcdt + (size_t)b * SEQLEN * D_IN_PROJ + D_INNER + ch;
  float acc = conv_b[ch];
#pragma unroll
  for (int k = 0; k < D_CONV; ++k) {
    int ls = l + k - D_CONV / 2;
    if (ls >= 0 && ls < SEQLEN)
      acc += src[(size_t)ls * D_IN_PROJ] * conv_w[ch * D_CONV + k];
  }
  convout[((size_t)b * SEQLEN + l) * CONV_DIM + ch] =
      acc / (1.0f + __expf(-acc));
}

// --------- K4: scatter fw/bw x2, B2, C2 (flip for bw), f32 -> bf16 ---------
__global__ __launch_bounds__(256) void scatter_kernel(
    const float* __restrict__ convout, bf16_t* __restrict__ x2,
    bf16_t* __restrict__ B2, bf16_t* __restrict__ C2) {
  const int bl = blockIdx.x;
  const int bb = bl / SEQLEN, l = bl % SEQLEN;
  const int b = bb % B_SZ;
  const bool bw = (bb >= B_SZ);
  const int lsrc = bw ? (SEQLEN - 1 - l) : l;
  const float* src = convout + ((size_t)b * SEQLEN + lsrc) * CONV_DIM;
  bf16_t* xd = x2 + (size_t)bl * D_INNER;
  for (int i = threadIdx.x; i < D_INNER; i += 256) xd[i] = (bf16_t)src[i];
  if (threadIdx.x < D_STATE) {
    int n = threadIdx.x;
    B2[(size_t)bl * D_STATE + n] =
        (bf16_t)src[D_INNER + (bw ? D_STATE + n : n)];
    C2[(size_t)bl * D_STATE + n] =
        (bf16_t)src[D_INNER + 2 * D_STATE + (bw ? D_STATE + n : n)];
  }
}

// -------------- K5: SSD intra-chunk (Ydiag) + chunk states -----------------
#define XLD 72   // padded LDS stride (144B rows -> 16B-aligned 32B runs)

__global__ __launch_bounds__(256) void ssd_diag_kernel(
    const bf16_t* __restrict__ x2, const bf16_t* __restrict__ B2,
    const bf16_t* __restrict__ C2, const float* __restrict__ dtb,
    const float* __restrict__ dAcs, float* __restrict__ y,
    float* __restrict__ states) {
  const int h = blockIdx.x, c = blockIdx.y, bbat = blockIdx.z;
  const int tid = threadIdx.x, wave = tid >> 5, lane = tid & 31;
  const int g = lane >> 4, r = lane & 15;
  const int t0 = bbat * SEQLEN + c * CHUNK;

  __shared__ __align__(16) bf16_t ldsXj[2][32][XLD];   // double-buffered
  __shared__ __align__(16) bf16_t ldsBj[2][32][XLD];
  __shared__ __align__(16) bf16_t ldsS[8][32][40];
  __shared__ float ldsDA[CHUNK];
  __shared__ float ldsDT[CHUNK];
  __shared__ float ldsW[CHUNK];

#if HAVE_TDM
  // Kick off DMA of j-block 0 while we stage dA/dt.
  if (wave == 0) {
    tdm_load_2d((unsigned)(size_t)(void*)&ldsXj[0][0][0],
                x2 + (size_t)t0 * D_INNER + h * HEADDIM,
                64u, 32u, (unsigned)D_INNER, 64u, 32u, 4u, 3u);
    tdm_load_2d((unsigned)(size_t)(void*)&ldsBj[0][0][0],
                B2 + (size_t)t0 * D_STATE,
                64u, 32u, (unsigned)D_STATE, 64u, 32u, 4u, 3u);
  }
#endif

  ldsDA[tid] = dAcs[(size_t)(t0 + tid) * NHEADS + h];
  ldsDT[tid] = dtb[(size_t)(t0 + tid) * NHEADS + h];
  __syncthreads();
  const float dal = ldsDA[CHUNK - 1];
  ldsW[tid] = __expf(dal - ldsDA[tid]) * ldsDT[tid];   // decay*dt

  const int iw = wave * 32;
  v16bf aC[2][2];
#pragma unroll
  for (int mi = 0; mi < 2; ++mi)
#pragma unroll
    for (int ks = 0; ks < 2; ++ks)
      aC[mi][ks] = load_a_frag(
          C2 + (size_t)(t0 + iw + mi * 16) * D_STATE + ks * 32, D_STATE, r, g);

  v8f Yacc[2][4];
#pragma unroll
  for (int mi = 0; mi < 2; ++mi)
#pragma unroll
    for (int nt = 0; nt < 4; ++nt) Yacc[mi][nt] = v8f_zero();
  v8f stAcc[2] = {v8f_zero(), v8f_zero()};
  const int mt = wave >> 1;
  const int ntbase = (wave & 1) * 2;

  for (int jb = 0; jb < 8; ++jb) {
    const int sb = jb & 1;
    __syncthreads();   // prior iteration's reads of ldsXj/ldsBj[sb] done
#if HAVE_TDM
    if (wave == 0) {
      __builtin_amdgcn_s_wait_tensorcnt((short)0);   // [sb] tiles landed
      if (jb < 7) {                                  // DMA next j-block
        tdm_load_2d((unsigned)(size_t)(void*)&ldsXj[sb ^ 1][0][0],
                    x2 + (size_t)(t0 + (jb + 1) * 32) * D_INNER + h * HEADDIM,
                    64u, 32u, (unsigned)D_INNER, 64u, 32u, 4u, 3u);
        tdm_load_2d((unsigned)(size_t)(void*)&ldsBj[sb ^ 1][0][0],
                    B2 + (size_t)(t0 + (jb + 1) * 32) * D_STATE,
                    64u, 32u, (unsigned)D_STATE, 64u, 32u, 4u, 3u);
      }
    }
#else
#pragma unroll
    for (int it = 0; it < 2; ++it) {
      int cid = tid + it * 256;          // 512 bf16x4 chunks
      int row = cid >> 4, c4 = cid & 15;
      *(bf16x4*)&ldsXj[sb][row][c4 * 4] = *(const bf16x4*)(
          x2 + (size_t)(t0 + jb * 32 + row) * D_INNER + h * HEADDIM + c4 * 4);
      *(bf16x4*)&ldsBj[sb][row][c4 * 4] = *(const bf16x4*)(
          B2 + (size_t)(t0 + jb * 32 + row) * D_STATE + c4 * 4);
    }
#endif
    __syncthreads();

    // states: A[p][s] = x[s][p] * decay*dt[s]; B[s][n].
    {
      v16bf ax;
      const int p = mt * 16 + r;
#pragma unroll
      for (int j = 0; j < 8; ++j) {
        int k0 = g * 16 + j, k1 = g * 16 + 8 + j;
        ax[j]     = (bf16_t)((float)ldsXj[sb][k0][p] * ldsW[jb * 32 + k0]);
        ax[8 + j] = (bf16_t)((float)ldsXj[sb][k1][p] * ldsW[jb * 32 + k1]);
      }
#pragma unroll
      for (int q = 0; q < 2; ++q) {
        v16bf bfr = load_b_frag_rowK(&ldsBj[sb][0][0], XLD,
                                     (ntbase + q) * 16 + r, g);
        stAcc[q] = wmma_bf16(ax, bfr, stAcc[q]);
      }
    }

    // Ydiag: causal -> only j-blocks at/below this wave's strip.
    if (jb <= wave) {
#pragma unroll
      for (int mi = 0; mi < 2; ++mi) {
#pragma unroll
        for (int jn = 0; jn < 2; ++jn) {
          v8f s = v8f_zero();
#pragma unroll
          for (int ks = 0; ks < 2; ++ks) {
            v16bf bfr = load_b_frag_rowN(&ldsBj[sb][jn * 16][ks * 32],
                                         XLD, r, g);
            s = wmma_bf16(aC[mi][ks], bfr, s);
          }
#pragma unroll
          for (int v = 0; v < 8; ++v) {
            int i = iw + mi * 16 + v + 8 * g;
            int j = jb * 32 + jn * 16 + r;
            float m = (j <= i) ? __expf(ldsDA[i] - ldsDA[j]) * ldsDT[j] : 0.0f;
            ldsS[wave][mi * 16 + v + 8 * g][jn * 16 + r] = (bf16_t)(s[v] * m);
          }
        }
      }
#pragma unroll
      for (int mi = 0; mi < 2; ++mi) {
        v16bf af = load_a_frag(&ldsS[wave][mi * 16][0], 40, r, g);
#pragma unroll
        for (int nt = 0; nt < 4; ++nt) {
          v16bf bx = load_b_frag_rowK(&ldsXj[sb][0][0], XLD, nt * 16 + r, g);
          Yacc[mi][nt] = wmma_bf16(af, bx, Yacc[mi][nt]);
        }
      }
    }
  }

  // Write Ydiag.
#pragma unroll
  for (int mi = 0; mi < 2; ++mi) {
    int i = iw + mi * 16 + 8 * g;
#pragma unroll
    for (int nt = 0; nt < 4; ++nt) {
      int p = nt * 16 + r;
#pragma unroll
      for (int v = 0; v < 8; ++v)
        y[(size_t)(t0 + i + v) * D_INNER + h * HEADDIM + p] = Yacc[mi][nt][v];
    }
  }
  // Write chunk states [bb][c][h][p][n] (f32: consumed by the scan).
  {
    float* st = states +
        ((((size_t)bbat * NCHUNKS + c) * NHEADS + h) * HEADDIM) * D_STATE;
    const int pbase = mt * 16 + 8 * g;
#pragma unroll
    for (int q = 0; q < 2; ++q) {
      int n = (ntbase + q) * 16 + r;
#pragma unroll
      for (int v = 0; v < 8; ++v)
        st[(size_t)(pbase + v) * D_STATE + n] = stAcc[q][v];
    }
  }
}

// --------------- K6: inter-chunk scan (f32 carry, bf16 out) ----------------
__global__ __launch_bounds__(256) void scan_kernel(
    const float* __restrict__ states, const float* __restrict__ dAlast,
    bf16_t* __restrict__ prevs) {
  const int blk = blockIdx.x;
  const int bb = blk / NHEADS, h = blk % NHEADS;
  float run[16];
#pragma unroll
  for (int i = 0; i < 16; ++i) run[i] = 0.0f;
  for (int c = 0; c < NCHUNKS; ++c) {
    float cd = __expf(dAlast[(bb * NCHUNKS + c) * NHEADS + h]);
    size_t base = (((size_t)bb * NCHUNKS + c) * NHEADS + h) *
                  (HEADDIM * D_STATE);
    const float* st = states + base;
    bf16_t* pv = prevs + base;
#pragma unroll
    for (int i = 0; i < 16; ++i) {
      int e = threadIdx.x + i * 256;
      pv[e] = (bf16_t)run[i];            // scan emits carry-before-update
      run[i] = cd * run[i] + st[e];
    }
  }
}

// ---------------- K7: Yoff = exp(dAcs) * (C @ prev^T), y += ----------------
__global__ __launch_bounds__(256) void ssd_yoff_kernel(
    const bf16_t* __restrict__ C2, const float* __restrict__ dAcs,
    const bf16_t* __restrict__ prevs, float* __restrict__ y) {
  const int h = blockIdx.x, c = blockIdx.y, bbat = blockIdx.z;
  const int tid = threadIdx.x, wave = tid >> 5, lane = tid & 31;
  const int g = lane >> 4, r = lane & 15;
  const int t0 = bbat * SEQLEN + c * CHUNK;

  __shared__ __align__(16) bf16_t ldsP[HEADDIM][XLD];   // prev[p][n]
  __shared__ float ldsDA[CHUNK];
  const bf16_t* pv = prevs +
      ((((size_t)bbat * NCHUNKS + c) * NHEADS + h) * HEADDIM) * D_STATE;
#if HAVE_TDM
  if (wave == 0)
    tdm_load_2d((unsigned)(size_t)(void*)&ldsP[0][0], pv,
                64u, 64u, (unsigned)D_STATE, 64u, 64u, 4u, 3u);
  ldsDA[tid] = dAcs[(size_t)(t0 + tid) * NHEADS + h];
  if (wave == 0) __builtin_amdgcn_s_wait_tensorcnt((short)0);
#else
  ldsDA[tid] = dAcs[(size_t)(t0 + tid) * NHEADS + h];
#pragma unroll
  for (int it = 0; it < 4; ++it) {
    int cid = tid + it * 256;            // 1024 bf16x4 chunks
    int p = cid >> 4, c4 = cid & 15;
    *(bf16x4*)&ldsP[p][c4 * 4] =
        *(const bf16x4*)(pv + (size_t)p * D_STATE + c4 * 4);
  }
#endif
  __syncthreads();

  const int iw = wave * 32;
  v16bf aC[2][2];
#pragma unroll
  for (int mi = 0; mi < 2; ++mi)
#pragma unroll
    for (int ks = 0; ks < 2; ++ks)
      aC[mi][ks] = load_a_frag(
          C2 + (size_t)(t0 + iw + mi * 16) * D_STATE + ks * 32, D_STATE, r, g);

  v8f acc[2][4];
#pragma unroll
  for (int mi = 0; mi < 2; ++mi)
#pragma unroll
    for (int nt = 0; nt < 4; ++nt) {
      acc[mi][nt] = v8f_zero();
#pragma unroll
      for (int ks = 0; ks < 2; ++ks) {
        v16bf bp = load_b_frag_rowN(&ldsP[nt * 16][ks * 32], XLD, r, g);
        acc[mi][nt] = wmma_bf16(aC[mi][ks], bp, acc[mi][nt]);
      }
    }

#pragma unroll
  for (int mi = 0; mi < 2; ++mi)
#pragma unroll
    for (int nt = 0; nt < 4; ++nt) {
      int p = nt * 16 + r;
#pragma unroll
      for (int v = 0; v < 8; ++v) {
        int i = iw + mi * 16 + v + 8 * g;
        size_t idx = (size_t)(t0 + i) * D_INNER + h * HEADDIM + p;
        y[idx] += acc[mi][nt][v] * __expf(ldsDA[i]);
      }
    }
}

// ------ K8: shift + fw/bw combine + Dterm + RMSNorm + SiLU(z) gating -------
__global__ __launch_bounds__(256) void combine_kernel(
    const float* __restrict__ y, const float* __restrict__ convout,
    const float* __restrict__ zxbcdt, const float* __restrict__ Dterm,
    const float* __restrict__ D_param, const float* __restrict__ norm_w,
    float* __restrict__ ynorm) {
  const int bl = blockIdx.x;
  const int b = bl / SEQLEN, l = bl % SEQLEN;
  __shared__ float red[8];
  const float* xog  = convout + (size_t)bl * CONV_DIM;
  const float* zrow = zxbcdt + (size_t)bl * D_IN_PROJ;
  float vals[8];
  float ss = 0.0f;
#pragma unroll
  for (int i = 0; i < 8; ++i) {
    int ch = threadIdx.x + i * 256;
    int h = ch / HEADDIM;
    float yfw = (l == 0) ? 0.0f
        : y[((size_t)b * SEQLEN + (l - 1)) * D_INNER + ch];
    float ybw = (l == SEQLEN - 1) ? 0.0f
        : y[((size_t)(B_SZ + b) * SEQLEN + (SEQLEN - 2 - l)) * D_INNER + ch];
    float dterm = Dterm[(size_t)bl * NHEADS + h] + D_param[h];
    float v = yfw + ybw + xog[ch] * dterm;
    vals[i] = v;
    ss += v * v;
  }
  const int lane = threadIdx.x & 31, wave = threadIdx.x >> 5;
#pragma unroll
  for (int off = 16; off > 0; off >>= 1) ss += __shfl_down(ss, off, 32);
  if (lane == 0) red[wave] = ss;
  __syncthreads();
  float tot = 0.0f;
#pragma unroll
  for (int w = 0; w < 8; ++w) tot += red[w];
  const float rms = rsqrtf(tot / (float)D_INNER + 1e-5f);
#pragma unroll
  for (int i = 0; i < 8; ++i) {
    int ch = threadIdx.x + i * 256;
    float z = zrow[ch];
    float sz = z / (1.0f + __expf(-z));
    ynorm[(size_t)bl * D_INNER + ch] = vals[i] * rms * norm_w[ch] * sz;
  }
}

// ------------------------------- launcher ----------------------------------

extern "C" void kernel_launch(void* const* d_in, const int* in_sizes, int n_in,
                              void* d_out, int out_size, void* d_ws,
                              size_t ws_size, hipStream_t stream) {
  (void)in_sizes; (void)n_in; (void)out_size; (void)ws_size;
  const float* u          = (const float*)d_in[0];
  const float* in_proj_w  = (const float*)d_in[1];
  const float* conv_w     = (const float*)d_in[2];
  const float* conv_b     = (const float*)d_in[3];
  const float* dt_bias    = (const float*)d_in[4];
  const float* A_log      = (const float*)d_in[5];
  const float* D_param    = (const float*)d_in[6];
  const float* fc_D_w     = (const float*)d_in[7];
  const float* norm_w     = (const float*)d_in[8];
  const float* out_proj_w = (const float*)d_in[9];
  float* out = (float*)d_out;

  // --- f32 workspace region (all offsets stay 16B aligned) ---
  float* wsf = (float*)d_ws;
  size_t off = 0;
  auto takef = [&](size_t n) { float* p = wsf + off; off += n; return p; };
  float* zxbcdt  = takef((size_t)B_SZ * SEQLEN * D_IN_PROJ);
  float* convout = takef((size_t)B_SZ * SEQLEN * CONV_DIM);
  float* dtbuf   = takef((size_t)BB * SEQLEN * NHEADS);
  float* dAcsbuf = takef((size_t)BB * SEQLEN * NHEADS);
  float* dAlast  = takef((size_t)BB * NCHUNKS * NHEADS);
  float* ybuf    = takef((size_t)BB * SEQLEN * D_INNER);
  float* stbuf   = takef((size_t)BB * NCHUNKS * NHEADS * HEADDIM * D_STATE);
  float* Dterm   = takef((size_t)MTOT * NHEADS);
  float* ynorm   = takef((size_t)B_SZ * SEQLEN * D_INNER);
  // --- bf16 workspace region ---
  bf16_t* wsh = (bf16_t*)(wsf + off);
  size_t hoff = 0;
  auto takeh = [&](size_t n) { bf16_t* p = wsh + hoff; hoff += n; return p; };
  bf16_t* x2      = takeh((size_t)BB * SEQLEN * D_INNER);
  bf16_t* Bbuf    = takeh((size_t)BB * SEQLEN * D_STATE);
  bf16_t* Cbuf    = takeh((size_t)BB * SEQLEN * D_STATE);
  bf16_t* prevbuf = takeh((size_t)BB * NCHUNKS * NHEADS * HEADDIM * D_STATE);
  bf16_t* wInBf   = takeh((size_t)D_IN_PROJ * D_MODEL);
  bf16_t* wOutBf  = takeh((size_t)D_MODEL * D_INNER);
  bf16_t* wDBf    = takeh((size_t)NHEADS * D_INNER);

  const dim3 blk(256);

  // 0. weight f32 -> bf16 converts
  cvt_bf16_kernel<<<dim3(4096), blk, 0, stream>>>(in_proj_w, wInBf,
                                                  D_IN_PROJ * D_MODEL);
  cvt_bf16_kernel<<<dim3(2048), blk, 0, stream>>>(out_proj_w, wOutBf,
                                                  D_MODEL * D_INNER);
  cvt_bf16_kernel<<<dim3(64), blk, 0, stream>>>(fc_D_w, wDBf,
                                                NHEADS * D_INNER);
  // 1. in_proj: zxbcdt = u @ in_proj_w^T
  gemm_wmma_kernel<<<dim3(D_IN_PROJ / GT_N, MTOT / GT_M), blk, 0, stream>>>(
      u, wInBf, zxbcdt, MTOT, D_IN_PROJ, D_MODEL, D_MODEL, D_MODEL, D_IN_PROJ);
  // 2. dt softplus + per-chunk cumsum
  dtprep_kernel<<<dim3(BB * NCHUNKS), dim3(32), 0, stream>>>(
      zxbcdt, dt_bias, A_log, dtbuf, dAcsbuf, dAlast);
  // 3. depthwise conv + SiLU
  dwconv_kernel<<<dim3(CONV_DIM / 256, B_SZ * SEQLEN), blk, 0, stream>>>(
      zxbcdt, conv_w, conv_b, convout);
  // 4. fw/bw scatter of x, B, C (bf16)
  scatter_kernel<<<dim3(BB * SEQLEN), blk, 0, stream>>>(convout, x2, Bbuf,
                                                        Cbuf);
  // 5. Dterm = x_og @ fc_D_w^T (x_og strided out of convout)
  gemm_wmma_kernel<<<dim3(1, MTOT / GT_M), blk, 0, stream>>>(
      convout, wDBf, Dterm, MTOT, NHEADS, D_INNER, CONV_DIM, D_INNER, NHEADS);
  // 6. SSD intra-chunk: Ydiag + chunk states
  ssd_diag_kernel<<<dim3(NHEADS, NCHUNKS, BB), blk, 0, stream>>>(
      x2, Bbuf, Cbuf, dtbuf, dAcsbuf, ybuf, stbuf);
  // 7. inter-chunk scan
  scan_kernel<<<dim3(BB * NHEADS), blk, 0, stream>>>(stbuf, dAlast, prevbuf);
  // 8. Yoff accumulate
  ssd_yoff_kernel<<<dim3(NHEADS, NCHUNKS, BB), blk, 0, stream>>>(
      Cbuf, dAcsbuf, prevbuf, ybuf);
  // 9. shift/combine/Dterm/RMSNorm/SiLU gate
  combine_kernel<<<dim3(B_SZ * SEQLEN), blk, 0, stream>>>(
      ybuf, convout, zxbcdt, Dterm, D_param, norm_w, ynorm);
  // 10. out_proj: out = ynorm @ out_proj_w^T
  gemm_wmma_kernel<<<dim3(D_MODEL / GT_N, MTOT / GT_M), blk, 0, stream>>>(
      ynorm, wOutBf, out, MTOT, D_MODEL, D_INNER, D_INNER, D_INNER, D_MODEL);
}
